// GNNModel_45208825758041
// MI455X (gfx1250) — matrix operand order repaired and checked by reading, not compiled, not run on hardware
//
#include <hip/hip_runtime.h>
#include <hip/hip_bf16.h>

typedef __attribute__((ext_vector_type(16))) _Float16 v16h;
typedef __attribute__((ext_vector_type(8)))  _Float16 v8h;
typedef __attribute__((ext_vector_type(8)))  float    v8f;
typedef __attribute__((ext_vector_type(4)))  float    v4f;

#define HID      64
#define NGRAPH   256
#define F_IN     320

// ---------------------------------------------------------------- utilities
__global__ void k_fill(float* __restrict__ p, float v, int n) {
    int i = blockIdx.x * blockDim.x + threadIdx.x;
    if (i < n) p[i] = v;
}

__global__ void k_deg_edges(const int* __restrict__ dst, float* __restrict__ deg, int E) {
    int e = blockIdx.x * blockDim.x + threadIdx.x;
    if (e < E) atomicAdd(&deg[dst[e]], 1.0f);
}

__global__ void k_dinv(const float* __restrict__ deg, float* __restrict__ dinv, int n) {
    int i = blockIdx.x * blockDim.x + threadIdx.x;
    if (i < n) {
        float d = deg[i];
        dinv[i] = (d > 0.0f) ? rsqrtf(d) : 0.0f;
    }
}

__global__ void k_norm(const int* __restrict__ src, const int* __restrict__ dst,
                       const float* __restrict__ dinv, float* __restrict__ norm, int E) {
    int e = blockIdx.x * blockDim.x + threadIdx.x;
    if (e < E) norm[e] = dinv[src[e]] * dinv[dst[e]];
}

// ---------------------------------------------------------------- WMMA GEMM
// C[M x 64] = A[M x K] * W[K x 64], A is f32 (layer 1, converted inline) or
// f16 (layers 2/3). One wave -> one 16-row M tile x full N=64.
// B is staged in LDS already swizzled into v_wmma_f32_16x16x32_f16 B-fragment
// layout: frag (kc,ntile), 32 lanes x 16 halves contiguous per lane.
template <typename AT, int K>
__global__ __launch_bounds__(256) void k_gemm(const AT* __restrict__ A,
                                              const float* __restrict__ W,
                                              float* __restrict__ C, int M) {
    constexpr int NFRAG = (K / 32) * 4;             // K-chunks * 4 N-tiles
    __shared__ __align__(64) _Float16 lbs[NFRAG * 512];

    const int tid = threadIdx.x;
    // stage + swizzle B: lane ln of frag holds B[kb + j + (j & 8)][n], j = 0..15
    for (int idx = tid; idx < NFRAG * 32; idx += 256) {
        int frag = idx >> 5, ln = idx & 31;
        int kb = ((frag >> 2) << 5) + ((ln & 16) ? 8 : 0);
        int n  = ((frag & 3) << 4) + (ln & 15);
        _Float16* o = &lbs[idx * 16];
#pragma unroll
        for (int j = 0; j < 16; ++j) o[j] = (_Float16)W[(kb + j + (j & 8)) * HID + n];
    }
    __syncthreads();

    const int wave = tid >> 5, lane = tid & 31;
    const int tile = blockIdx.x * 8 + wave;
    const int mt = (M + 15) >> 4;
    if (tile >= mt) return;                          // wave-uniform: EXEC stays all-1s

    const int arow = (tile << 4) + (lane & 15);      // A row this lane feeds
    const int kq = (lane >> 4) << 3;                 // 0 or 8 (A-frag K phase)

    v8f acc0 = {}, acc1 = {}, acc2 = {}, acc3 = {};

    for (int kc = 0; kc < K; kc += 32) {
        // ---- A fragment (16-bit A 16x32 layout, ISA 7.12.2)
        v16h a;
        const AT* ap = A + (size_t)arow * K + kc + kq;
        if constexpr (sizeof(AT) == 4) {             // f32 source: convert inline
            const v4f* f = (const v4f*)ap;
            v4f u0 = f[0], u1 = f[1], u2 = f[4], u3 = f[5];
#pragma unroll
            for (int j = 0; j < 4; ++j) {
                a[j]      = (_Float16)u0[j];
                a[4 + j]  = (_Float16)u1[j];
                a[8 + j]  = (_Float16)u2[j];
                a[12 + j] = (_Float16)u3[j];
            }
        } else {                                     // f16 source
            v8h h0 = *(const v8h*)ap;
            v8h h1 = *(const v8h*)(ap + 16);
#pragma unroll
            for (int j = 0; j < 8; ++j) { a[j] = h0[j]; a[8 + j] = h1[j]; }
        }
        // ---- B fragments from LDS (pre-swizzled, contiguous per lane)
        const _Float16* bp = &lbs[((kc >> 5) * 4) * 512 + lane * 16];
        v16h b0 = *(const v16h*)(bp);
        v16h b1 = *(const v16h*)(bp + 512);
        v16h b2 = *(const v16h*)(bp + 1024);
        v16h b3 = *(const v16h*)(bp + 1536);

        acc0 = __builtin_amdgcn_wmma_f32_16x16x32_f16(false, a, false, b0, (short)0, acc0, false, false);
        acc1 = __builtin_amdgcn_wmma_f32_16x16x32_f16(false, a, false, b1, (short)0, acc1, false, false);
        acc2 = __builtin_amdgcn_wmma_f32_16x16x32_f16(false, a, false, b2, (short)0, acc2, false, false);
        acc3 = __builtin_amdgcn_wmma_f32_16x16x32_f16(false, a, false, b3, (short)0, acc3, false, false);
    }

    // ---- store C (32-bit C/D layout: VGPR r -> M = r + 8*(lane>=16), N = lane&15)
    const int crow0 = (tile << 4) + ((lane >> 4) << 3);
    const int ccol = lane & 15;
#pragma unroll
    for (int r = 0; r < 8; ++r) {
        float* cp = C + (size_t)(crow0 + r) * HID + ccol;
        cp[0]  = acc0[r];
        cp[16] = acc1[r];
        cp[32] = acc2[r];
        cp[48] = acc3[r];
    }
}

// ---------------------------------------------------------------- aggregation
// agg[dst] += h[src] * norm[e]; 16 threads per edge, float4 each. h/agg are
// L2-resident (25.6 MB each on a 192 MB L2), so these atomics stay on-chip.
__global__ __launch_bounds__(256) void k_agg(const int* __restrict__ src,
                                             const int* __restrict__ dst,
                                             const float* __restrict__ norm,
                                             const float* __restrict__ h,
                                             float* __restrict__ agg, int E) {
    int gid = blockIdx.x * blockDim.x + threadIdx.x;
    int e = gid >> 4;
    if (e >= E) return;
    int q = gid & 15;
    float w = norm[e];
    int s = src[e], d = dst[e];
    v4f v = *(const v4f*)(h + (size_t)s * HID + q * 4);
    float* ap = agg + (size_t)d * HID + q * 4;
    atomicAdd(ap + 0, v[0] * w);
    atomicAdd(ap + 1, v[1] * w);
    atomicAdd(ap + 2, v[2] * w);
    atomicAdd(ap + 3, v[3] * w);
}

// finalize: fold self-loop (h[i]*dinv[i]^2) + bias (+ ReLU), emit f16 for next
// GEMM and/or f32 (layer 3, in place over h).
__global__ __launch_bounds__(256) void k_fin(const float* __restrict__ agg,
                                             const float* __restrict__ h,
                                             const float* __restrict__ dinv,
                                             const float* __restrict__ bias,
                                             float* __restrict__ out32,
                                             _Float16* __restrict__ out16,
                                             int relu, int n) {
    int i = blockIdx.x * blockDim.x + threadIdx.x;
    if (i >= n * HID) return;
    int node = i >> 6, f = i & 63;
    float dv = dinv[node];
    float v = agg[i] + h[i] * dv * dv + bias[f];
    if (relu) v = fmaxf(v, 0.0f);
    if (out32) out32[i] = v;
    if (out16) out16[i] = (_Float16)v;
}

// ---------------------------------------------------------------- pooling/head
__global__ __launch_bounds__(256) void k_pool(const float* __restrict__ h,
                                              const int* __restrict__ batch,
                                              float* __restrict__ pool,
                                              float* __restrict__ cnt, int n) {
    int gid = blockIdx.x * blockDim.x + threadIdx.x;
    int node = gid >> 4;
    if (node >= n) return;
    int q = gid & 15;
    int g = batch[node];
    v4f v = *(const v4f*)(h + (size_t)node * HID + q * 4);
    float* pp = pool + (size_t)g * HID + q * 4;
    atomicAdd(pp + 0, v[0]);
    atomicAdd(pp + 1, v[1]);
    atomicAdd(pp + 2, v[2]);
    atomicAdd(pp + 3, v[3]);
    if (q == 0) atomicAdd(&cnt[g], 1.0f);
}

__global__ void k_head(const float* __restrict__ pool, const float* __restrict__ cnt,
                       const float* __restrict__ Wl, const float* __restrict__ bl,
                       float* __restrict__ out) {
    int g = threadIdx.x;
    if (g >= NGRAPH) return;
    float c = fmaxf(cnt[g], 1.0f);
    float s = 0.0f;
#pragma unroll
    for (int f = 0; f < HID; ++f) s += pool[g * HID + f] * Wl[f];
    out[g] = s / c + bl[0];
}

// ---------------------------------------------------------------- launcher
extern "C" void kernel_launch(void* const* d_in, const int* in_sizes, int n_in,
                              void* d_out, int out_size, void* d_ws, size_t ws_size,
                              hipStream_t stream) {
    const float* x   = (const float*)d_in[0];
    const int*   ei  = (const int*)d_in[1];        // [2, E] int32
    const int*   bat = (const int*)d_in[2];
    const float* W1  = (const float*)d_in[3];
    const float* b1  = (const float*)d_in[4];
    const float* W2  = (const float*)d_in[5];
    const float* b2  = (const float*)d_in[6];
    const float* W3  = (const float*)d_in[7];
    const float* b3  = (const float*)d_in[8];
    const float* Wl  = (const float*)d_in[9];
    const float* bl  = (const float*)d_in[10];
    float* out = (float*)d_out;

    const int N = in_sizes[2];
    const int E = in_sizes[1] / 2;
    const int* src = ei;
    const int* dst = ei + E;

    // carve workspace (256-byte aligned slabs)
    char* w = (char*)d_ws;
    auto carve = [&](size_t bytes) {
        void* p = (void*)w;
        w += (bytes + 255) & ~(size_t)255;
        return p;
    };
    float*    deg  = (float*)carve((size_t)N * 4);
    float*    dinv = (float*)carve((size_t)N * 4);
    float*    norm = (float*)carve((size_t)E * 4);
    float*    h    = (float*)carve((size_t)N * HID * 4);
    float*    agg  = (float*)carve((size_t)N * HID * 4);
    float*    pool = (float*)carve((size_t)NGRAPH * HID * 4);
    float*    cnt  = (float*)carve((size_t)NGRAPH * 4);
    _Float16* acth = (_Float16*)carve((size_t)N * HID * 2);

    const int TB = 256;
    auto blk = [](long long n, int tb) { return (int)((n + tb - 1) / tb); };
    const int gemm_blocks = blk(((N + 15) / 16), 8);

    // --- normalization: deg (self-loop => init 1), dinv, per-edge norm
    k_fill<<<blk(N, TB), TB, 0, stream>>>(deg, 1.0f, N);
    k_deg_edges<<<blk(E, TB), TB, 0, stream>>>(dst, deg, E);
    k_dinv<<<blk(N, TB), TB, 0, stream>>>(deg, dinv, N);
    k_norm<<<blk(E, TB), TB, 0, stream>>>(src, dst, dinv, norm, E);

    // --- layer 1: h = x @ W1 ; aggregate ; +self-loop +bias, ReLU -> acth (f16)
    k_gemm<float, F_IN><<<gemm_blocks, TB, 0, stream>>>(x, W1, h, N);
    k_fill<<<blk((long long)N * HID, TB), TB, 0, stream>>>(agg, 0.0f, N * HID);
    k_agg<<<blk((long long)E * 16, TB), TB, 0, stream>>>(src, dst, norm, h, agg, E);
    k_fin<<<blk((long long)N * HID, TB), TB, 0, stream>>>(agg, h, dinv, b1, nullptr, acth, 1, N);

    // --- layer 2
    k_gemm<_Float16, HID><<<gemm_blocks, TB, 0, stream>>>(acth, W2, h, N);
    k_fill<<<blk((long long)N * HID, TB), TB, 0, stream>>>(agg, 0.0f, N * HID);
    k_agg<<<blk((long long)E * 16, TB), TB, 0, stream>>>(src, dst, norm, h, agg, E);
    k_fin<<<blk((long long)N * HID, TB), TB, 0, stream>>>(agg, h, dinv, b2, nullptr, acth, 1, N);

    // --- layer 3 (no ReLU, f32 result written in place over h)
    k_gemm<_Float16, HID><<<gemm_blocks, TB, 0, stream>>>(acth, W3, h, N);
    k_fill<<<blk((long long)N * HID, TB), TB, 0, stream>>>(agg, 0.0f, N * HID);
    k_agg<<<blk((long long)E * 16, TB), TB, 0, stream>>>(src, dst, norm, h, agg, E);
    k_fin<<<blk((long long)N * HID, TB), TB, 0, stream>>>(agg, h, dinv, b3, h, nullptr, 0, N);

    // --- mean pool + linear head
    k_fill<<<blk(NGRAPH * HID, TB), TB, 0, stream>>>(pool, 0.0f, NGRAPH * HID);
    k_fill<<<1, TB, 0, stream>>>(cnt, 0.0f, NGRAPH);
    k_pool<<<blk((long long)N * 16, TB), TB, 0, stream>>>(h, bat, pool, cnt, N);
    k_head<<<1, TB, 0, stream>>>(pool, cnt, Wl, bl, out);
}
